// DecoderRNN_34248069219140
// MI455X (gfx1250) — matrix operand order repaired and checked by reading, not compiled, not run on hardware
//
#include <hip/hip_runtime.h>
#include <hip/hip_bf16.h>
#include <math.h>

// ---------------------------------------------------------------------------
// Pointer-network decoder for MI455X (gfx1250, wave32).
// Persistent-kernel design: 32 workgroups x 256 threads (== batch size), only
// THREE grid barriers per decode step. Recurrent GEMM uses
// v_wmma_f32_16x16x32_bf16 with a bf16 hi/lo split (3 WMMAs per K-block)
// for ~fp32 accuracy. Workgroup b owns batch row b for the LSTM pointwise
// phase, so dec_proj needs only a local __syncthreads().
// ---------------------------------------------------------------------------

typedef __bf16 bf16_t;
typedef __attribute__((ext_vector_type(16))) __bf16 v16bf;
typedef __attribute__((ext_vector_type(8)))  __bf16 v8bf;
typedef __attribute__((ext_vector_type(8)))  float  v8f;

#define HSZ   512      // hidden size
#define BSZ   32       // batch
#define NSEQ  512      // encoder positions == decode steps
#define ATTD  32       // attention dim
#define G4H   2048     // 4*H
#define NWG   32       // persistent workgroups (must equal BSZ)
#define TPB   256      // threads per workgroup (8 waves)
#define NTHR  (NWG*TPB)

// ---- workspace layout (bytes) ---------------------------------------------
#define WS_WHH_HI   0                                  // 2048*512 bf16 = 2 MB
#define WS_WHH_LO   (WS_WHH_HI + G4H*HSZ*2)            // 2 MB
#define WS_ENCPROJ  (WS_WHH_LO + G4H*HSZ*2)            // 16384*32 f32 = 2 MB
#define WS_GATES    (WS_ENCPROJ + BSZ*NSEQ*ATTD*4)     // 32*2048 f32 = 256 KB
#define WS_HBF_HI   (WS_GATES + BSZ*G4H*4)             // 32*512 bf16
#define WS_HBF_LO   (WS_HBF_HI + BSZ*HSZ*2)
#define WS_CCUR     (WS_HBF_LO + BSZ*HSZ*2)            // 32*512 f32
#define WS_DEC      (WS_CCUR + BSZ*HSZ*4)              // 32*32 f32
#define WS_ARG      (WS_DEC + BSZ*ATTD*4)              // 32 u64
#define WS_BAR      (WS_ARG + BSZ*8)                   // 2 u32

__device__ __forceinline__ float sigm(float x) { return 1.0f / (1.0f + __expf(-x)); }

__device__ __forceinline__ unsigned ord_f32(float f) {
    unsigned u = __float_as_uint(f);
    return (u & 0x80000000u) ? ~u : (u | 0x80000000u);
}

// grid-wide sense barrier (all NWG workgroups are resident)
__device__ __forceinline__ void grid_barrier(volatile unsigned* cnt, volatile unsigned* gen) {
    __syncthreads();
    if (threadIdx.x == 0) {
        __threadfence();
        unsigned g = *gen;
        if (atomicAdd((unsigned*)cnt, 1u) == NWG - 1u) {
            *cnt = 0;
            __threadfence();
            atomicAdd((unsigned*)gen, 1u);
        } else {
            while (*gen == g) { __builtin_amdgcn_s_sleep(1); }
        }
        __threadfence();
    }
    __syncthreads();
}

// A-matrix fragment (16x32 bf16, contiguous K order matching B):
//  lanes 0-15 : a[0..15]=K 0..15  (of this 32-K block)
//  lanes16-31 : a[0..15]=K16..31
// p points at row*512 + kk*32 + ((lane&16)?16:0)
__device__ __forceinline__ v16bf load_a_frag(const bf16_t* p) {
    v8bf lo = *(const v8bf*)(p);
    v8bf hi = *(const v8bf*)(p + 8);
    v16bf r;
#pragma unroll
    for (int j = 0; j < 8; ++j) { r[j] = lo[j]; r[8 + j] = hi[j]; }
    return r;
}

// ---------------------------------------------------------------------------
// Kernel 1: prep — split w_hh into bf16 hi/lo, init h/c state, reset barriers.
// Re-run safe (graph replay): resets barrier counters and argmax keys.
// ---------------------------------------------------------------------------
__global__ void prep_kernel(const float* __restrict__ w_hh,
                            const float* __restrict__ enc_h,
                            const float* __restrict__ enc_c,
                            char* __restrict__ ws) {
    bf16_t* whh_hi = (bf16_t*)(ws + WS_WHH_HI);
    bf16_t* whh_lo = (bf16_t*)(ws + WS_WHH_LO);
    bf16_t* hbf_hi = (bf16_t*)(ws + WS_HBF_HI);
    bf16_t* hbf_lo = (bf16_t*)(ws + WS_HBF_LO);
    float*  ccur   = (float*)(ws + WS_CCUR);
    unsigned long long* argb = (unsigned long long*)(ws + WS_ARG);
    unsigned* bar = (unsigned*)(ws + WS_BAR);

    unsigned gid = blockIdx.x * blockDim.x + threadIdx.x;
    if (gid < (unsigned)(G4H * HSZ)) {
        float w = w_hh[gid];
        bf16_t hi = (bf16_t)w;
        whh_hi[gid] = hi;
        whh_lo[gid] = (bf16_t)(w - (float)hi);
    }
    if (gid < (unsigned)(BSZ * HSZ)) {
        float h = enc_h[gid];            // encoder_hidden[0] flat [B,H]
        bf16_t hh = (bf16_t)h;
        hbf_hi[gid] = hh;
        hbf_lo[gid] = (bf16_t)(h - (float)hh);
        ccur[gid] = enc_c[gid];
    }
    if (gid < (unsigned)BSZ) argb[gid] = 0ull;
    if (gid == 0) { bar[0] = 0u; bar[1] = 0u; }
}

// ---------------------------------------------------------------------------
// Kernel 2: enc_proj[b,n,a] = dot(encoder_outputs[b,n,:], w1[a,:]) + b1[a]
// one block per (b,n) row; fp32 exact.
// ---------------------------------------------------------------------------
__global__ void encproj_kernel(const float* __restrict__ enc,
                               const float* __restrict__ w1,
                               const float* __restrict__ b1,
                               char* __restrict__ ws) {
    __shared__ float row[HSZ];
    __shared__ float red[TPB];
    float* encproj = (float*)(ws + WS_ENCPROJ);

    unsigned bn = blockIdx.x;                 // 0..16383
    const float* src = enc + (size_t)bn * HSZ;
    for (int k = threadIdx.x; k < HSZ; k += TPB) row[k] = src[k];
    __syncthreads();

    int a = threadIdx.x & 31;                 // attention dim
    int seg = threadIdx.x >> 5;               // 0..7
    const float* wr = w1 + (size_t)a * HSZ + seg * 64;
    const float* rr = row + seg * 64;
    float p = 0.f;
#pragma unroll
    for (int k = 0; k < 64; ++k) p += rr[k] * wr[k];
    red[threadIdx.x] = p;
    __syncthreads();
    if (threadIdx.x < 32) {
        float s = b1[threadIdx.x];
#pragma unroll
        for (int q = 0; q < 8; ++q) s += red[threadIdx.x + q * 32];
        encproj[(size_t)bn * ATTD + threadIdx.x] = s;
    }
}

// ---------------------------------------------------------------------------
// Kernel 3: persistent sequential decoder (512 steps, 3 grid barriers/step).
// ---------------------------------------------------------------------------
__global__ void __launch_bounds__(TPB, 1)
decoder_kernel(const float* __restrict__ inco,     // [B,N,2]
               const float* __restrict__ w_ih,     // [2048,2]
               const float* __restrict__ b_ih,
               const float* __restrict__ b_hh,
               const float* __restrict__ w2,       // [32,512]
               const float* __restrict__ b2,
               const float* __restrict__ vvec,     // [32]
               const float* __restrict__ bv,
               char* __restrict__ ws,
               float* __restrict__ dout) {         // [B,N,steps]
    const bf16_t* whh_hi = (const bf16_t*)(ws + WS_WHH_HI);
    const bf16_t* whh_lo = (const bf16_t*)(ws + WS_WHH_LO);
    const float*  encproj = (const float*)(ws + WS_ENCPROJ);
    float*  gates  = (float*)(ws + WS_GATES);
    bf16_t* hbf_hi = (bf16_t*)(ws + WS_HBF_HI);
    bf16_t* hbf_lo = (bf16_t*)(ws + WS_HBF_LO);
    float*  ccur   = (float*)(ws + WS_CCUR);
    float*  dec    = (float*)(ws + WS_DEC);
    unsigned long long* argb = (unsigned long long*)(ws + WS_ARG);
    volatile unsigned* bar = (unsigned*)(ws + WS_BAR);

    __shared__ float xsh[BSZ][2];   // current input coordinates (all batches)
    __shared__ float hsh[HSZ];      // this WG's batch h-row (post-LSTM)

    const int tid  = threadIdx.x;
    const int lane = tid & 31;
    const int wgid = blockIdx.x * 8 + (tid >> 5);     // global wave id 0..255
    const unsigned gtid = blockIdx.x * TPB + tid;     // 0..8191
    const int bown = blockIdx.x;                      // batch row owned in phase B

    // START token x0 = 0
    if (tid < BSZ) { xsh[tid][0] = 0.f; xsh[tid][1] = 0.f; }
    __syncthreads();

    // per-wave tile assignment for the recurrent GEMM (2 x 128 tiles of 16x16)
    const int mtile = wgid >> 7;          // 0..1
    const int ntile = wgid & 127;         // 0..127
    const int rowM  = mtile * 16 + (lane & 15);      // batch row for A
    const int nglob = ntile * 16 + (lane & 15);      // gate column for B
    const int koffA = (lane & 16) ? 16 : 0;
    const int koffB = (lane & 16) ? 16 : 0;
    const float wih0 = w_ih[nglob * 2 + 0];
    const float wih1 = w_ih[nglob * 2 + 1];
    const float bsum = b_ih[nglob] + b_hh[nglob];

    for (int t = 0; t < NSEQ; ++t) {
        // ---------- Phase A: gates = h @ w_hh.T + x @ w_ih.T + b (WMMA) ----
        {
            v8f acc = {0.f, 0.f, 0.f, 0.f, 0.f, 0.f, 0.f, 0.f};
            const bf16_t* ah = hbf_hi + (size_t)rowM * HSZ;
            const bf16_t* al = hbf_lo + (size_t)rowM * HSZ;
            const bf16_t* bh = whh_hi + (size_t)nglob * HSZ;
            const bf16_t* bl = whh_lo + (size_t)nglob * HSZ;
#pragma unroll
            for (int kk = 0; kk < 16; ++kk) {
                const int ka = kk * 32 + koffA;
                const int kb = kk * 32 + koffB;
                v16bf a_hi = load_a_frag(ah + ka);
                v16bf a_lo = load_a_frag(al + ka);
                v16bf b_hi = *(const v16bf*)(bh + kb);
                v16bf b_lo = *(const v16bf*)(bl + kb);
                acc = __builtin_amdgcn_wmma_f32_16x16x32_bf16(false, a_hi, false, b_hi,
                                                              (short)0, acc, false, false);
                acc = __builtin_amdgcn_wmma_f32_16x16x32_bf16(false, a_hi, false, b_lo,
                                                              (short)0, acc, false, false);
                acc = __builtin_amdgcn_wmma_f32_16x16x32_bf16(false, a_lo, false, b_hi,
                                                              (short)0, acc, false, false);
            }
#pragma unroll
            for (int r = 0; r < 8; ++r) {
                int m = mtile * 16 + r + ((lane & 16) ? 8 : 0);
                float val = acc[r] + xsh[m][0] * wih0 + xsh[m][1] * wih1 + bsum;
                gates[(size_t)m * G4H + nglob] = val;
            }
        }
        grid_barrier(&bar[0], &bar[1]);    // barrier 1: gates complete

        // ---------- Phase B: LSTM pointwise for batch row bown (WG-local) --
        {
            const float* g = gates + (size_t)bown * G4H;
            for (int j = tid; j < HSZ; j += TPB) {
                unsigned u = (unsigned)bown * HSZ + j;
                float gi = g[j];
                float gf = g[HSZ + j];
                float gg = g[2 * HSZ + j];
                float go = g[3 * HSZ + j];
                float cn = sigm(gf) * ccur[u] + sigm(gi) * tanhf(gg);
                float hn = sigm(go) * tanhf(cn);
                ccur[u] = cn;
                bf16_t hh = (bf16_t)hn;
                hbf_hi[u] = hh;
                hbf_lo[u] = (bf16_t)(hn - (float)hh);
                hsh[j] = hn;               // fp32 h row stays in LDS
            }
        }
        __syncthreads();                    // LDS h-row ready (no grid barrier)

        // ---------- Phase C1: dec_proj[bown,:] = h @ w2.T + b2 (from LDS) --
        {
            int aa0 = (tid >> 5) * 4;       // 4 attention dims per wave
#pragma unroll
            for (int q = 0; q < 4; ++q) {
                int aa = aa0 + q;
                const float* wr = w2 + (size_t)aa * HSZ;
                float p = 0.f;
                for (int k = lane; k < HSZ; k += 32) p += hsh[k] * wr[k];
#pragma unroll
                for (int off = 16; off > 0; off >>= 1) p += __shfl_xor(p, off, 32);
                if (lane == 0) dec[bown * ATTD + aa] = p + b2[aa];
            }
        }
        grid_barrier(&bar[0], &bar[1]);    // barrier 2: h(bf16) + dec complete

        // ---------- Phase C2: logits + packed-key argmax -------------------
        for (unsigned u = gtid; u < (unsigned)(BSZ * NSEQ); u += NTHR) {
            unsigned b = u >> 9, n = u & 511u;
            const float* ep = encproj + (size_t)u * ATTD;
            const float* dp = dec + b * ATTD;
            float s = bv[0];
#pragma unroll
            for (int a = 0; a < ATTD; ++a) s += vvec[a] * tanhf(ep[a] + dp[a]);
            dout[(size_t)u * NSEQ + t] = s;    // out[b, n, t]
            unsigned long long key = ((unsigned long long)(t + 1) << 41)
                                   | ((unsigned long long)ord_f32(s) << 9)
                                   | (unsigned long long)(511u - n);
            atomicMax(&argb[b], key);
        }
        grid_barrier(&bar[0], &bar[1]);    // barrier 3: argmax complete

        // ---------- Phase D: each WG reads argmax, fetches next x into LDS -
        if (tid < BSZ) {
            unsigned long long key = argb[tid];
            unsigned n = 511u - (unsigned)(key & 0x1FFull);
            const float* ic = inco + ((size_t)tid * NSEQ + n) * 2;
            xsh[tid][0] = ic[0];
            xsh[tid][1] = ic[1];
        }
        __syncthreads();
    }
}

// ---------------------------------------------------------------------------
extern "C" void kernel_launch(void* const* d_in, const int* in_sizes, int n_in,
                              void* d_out, int out_size, void* d_ws, size_t ws_size,
                              hipStream_t stream) {
    const float* inco   = (const float*)d_in[0];   // [32,512,2]
    const float* encout = (const float*)d_in[1];   // [32,512,512]
    const float* ench   = (const float*)d_in[2];   // [1,32,512]
    const float* encc   = (const float*)d_in[3];   // [1,32,512]
    const float* w_ih   = (const float*)d_in[4];   // [2048,2]
    const float* w_hh   = (const float*)d_in[5];   // [2048,512]
    const float* b_ih   = (const float*)d_in[6];
    const float* b_hh   = (const float*)d_in[7];
    const float* w1     = (const float*)d_in[8];   // [32,512]
    const float* b1     = (const float*)d_in[9];
    const float* w2     = (const float*)d_in[10];  // [32,512]
    const float* b2     = (const float*)d_in[11];
    const float* vvec   = (const float*)d_in[12];  // [1,32]
    const float* bv     = (const float*)d_in[13];
    float* out = (float*)d_out;
    char* ws = (char*)d_ws;

    (void)in_sizes; (void)n_in; (void)out_size; (void)ws_size;

    // 1) prep: split weights, init state, reset barrier/argmax (re-run safe)
    prep_kernel<<<dim3((G4H * HSZ + TPB - 1) / TPB), dim3(TPB), 0, stream>>>(
        w_hh, ench, encc, ws);

    // 2) encoder attention projection (one-time GEMM, fp32 exact)
    encproj_kernel<<<dim3(BSZ * NSEQ), dim3(TPB), 0, stream>>>(encout, w1, b1, ws);

    // 3) persistent sequential decoder with WMMA recurrent GEMM
    decoder_kernel<<<dim3(NWG), dim3(TPB), 0, stream>>>(
        inco, w_ih, b_ih, b_hh, w2, b2, vvec, bv, ws, out);
}
